// S4_58944131170545
// MI455X (gfx1250) — matrix operand order repaired and checked by reading, not compiled, not run on hardware
//
#include <hip/hip_runtime.h>
#include <hip/hip_bf16.h>

// S4 causal convolution via chunked parallel scan + WMMA on gfx1250 (MI455X).
// B=8, L=2048, D=768, N=64, chunk Lc=64, 32 chunks.
//
// Roofline: minimal traffic ~100MB (u in + y out) -> ~4.5us @ 23.3TB/s.
// Chunked scan makes compute ~10 GFLOP (vs 26 TFLOP for dense Toeplitz GEMM),
// so the whole problem is memory-bound; the per-chunk GEMMs map onto
// v_wmma_f32_16x16x32_f16 (f16 operands, f32 accumulate), and u-chunk staging
// uses gfx1250 async global->LDS copies (ASYNCcnt) double-buffered across the
// serial chunk loop.

typedef _Float16 f16;
typedef __attribute__((ext_vector_type(16))) _Float16 v16h;
typedef __attribute__((ext_vector_type(8)))  float    v8f;
typedef __attribute__((ext_vector_type(4)))  int      v4i;

#define BATCH 8
#define SEQL  2048
#define DCH   768
#define NST   64
#define LC    64
#define NCH   32    // SEQL / LC
#define KDIM  192   // LC + 2*NST

// Async global->LDS (gfx1250). Guarded: falls back to plain LDS staging if the
// toolchain lacks the builtin (also keeps the host pass clean).
#ifdef __has_builtin
#if __has_builtin(__builtin_amdgcn_global_load_async_to_lds_b128)
#define HAVE_ASYNC_LDS 1
#endif
#endif
#ifndef HAVE_ASYNC_LDS
#define HAVE_ASYNC_LDS 0
#endif

#if HAVE_ASYNC_LDS
// Param 0 (probe-diagnosed): non-const int4* in global AS; param 1: LDS int4*.
typedef __attribute__((address_space(1))) v4i* gv4_t;
typedef __attribute__((address_space(3))) v4i* lv4_t;
__device__ inline void async_copy_1k(const f16* gsrc, f16* ldst, int lane) {
  // 1KB chunk: each of 32 lanes copies 16B at lane*16, then +512 via imm offset.
  gv4_t g = (gv4_t)(uintptr_t)(gsrc + lane * 8);
  lv4_t l = (lv4_t)(unsigned)(uintptr_t)(ldst + lane * 8);
  __builtin_amdgcn_global_load_async_to_lds_b128(g, l, 0, 0);
  __builtin_amdgcn_global_load_async_to_lds_b128(g, l, 512, 0);
}
__device__ inline void wait_async0() {
#if __has_builtin(__builtin_amdgcn_s_wait_asynccnt)
  __builtin_amdgcn_s_wait_asynccnt(0);
#else
  asm volatile("s_wait_asynccnt 0x0" ::: "memory");
#endif
}
#endif

// ---- WMMA fragment loaders (contiguous k order: k = 16h + e per lane half, same for A and B, wave32) ----
// A (16x32 f16): lane m = lane&15, h = lane>>4;
//                elems 0..15 -> k = 32*ks + 16h + e   (one contiguous 32B run)
__device__ inline v16h load_frag_a(const f16* base, int row_stride, int mtile, int ks, int lane) {
  const int m = mtile * 16 + (lane & 15);
  const int h = lane >> 4;
  const f16* r = base + m * row_stride + ks * 32 + h * 16;
  v16h a;
#pragma unroll
  for (int e = 0; e < 16; ++e) a[e] = r[e];
  return a;
}
// B (32x16 f16), LDS as [n][k]: lane n = lane&15, h = lane>>4;
// elems 0..15 -> k = 32*ks + 16h + e   (one contiguous 32B run)
__device__ inline v16h load_frag_b(const f16* base, int row_stride, int ks, int lane) {
  const int n = lane & 15;
  const int h = lane >> 4;
  const f16* r = base + n * row_stride + ks * 32 + h * 16;
  v16h b;
#pragma unroll
  for (int e = 0; e < 16; ++e) b[e] = r[e];
  return b;
}

// ---- Kernel 0: u (B,L,D) f32 -> u16 [d][chunk][b][64] f16 ----
// Chunk-blocked layout makes each (d, chunk) operand a contiguous 1KB block,
// ready for a linear async DMA straight into the LDS B-operand buffer.
__global__ __launch_bounds__(256) void s4_transpose_u(const float* __restrict__ u,
                                                      f16* __restrict__ u16) {
  __shared__ f16 tile[32 * 32 * 8];            // [j(d)][i(l)][b], 16 KB
  const int l0 = blockIdx.x * 32;
  const int d0 = blockIdx.y * 32;
  const int tid = threadIdx.x;
#pragma unroll
  for (int b = 0; b < 8; ++b) {
#pragma unroll
    for (int rep = 0; rep < 4; ++rep) {
      int idx = rep * 256 + tid;
      int i = idx >> 5, j = idx & 31;        // consecutive tid -> consecutive d: coalesced
      float v = u[((size_t)b * SEQL + (l0 + i)) * DCH + d0 + j];
      tile[(j * 32 + i) * 8 + b] = (f16)v;
    }
  }
  __syncthreads();
  const int I = l0 >> 6, k0 = l0 & 63;       // chunk id, k offset within chunk
  const int j = tid >> 3, bb = tid & 7;
  f16* dst = u16 + ((((size_t)(d0 + j) * NCH + I) * 8 + bb) * 64 + k0);
#pragma unroll
  for (int i = 0; i < 32; ++i)               // 64B contiguous store per thread
    dst[i] = tile[(j * 32 + i) * 8 + bb];
}

// ---- Kernel 1: per-d operator precompute ----
// Scat [d][128][64] f16 : rows 0..63 Re(lam^{63-t} * Bbar), rows 64..127 Im(...)
// Mcat [d][64][192] f16 : cols 0..63 T0[t][s]=K[t-s] (lower-tri),
//                         cols 64..127 Re(C*lam^{t+1}), cols 128..191 -Im(C*lam^{t+1})
// lam64 [d][64][2] f32  : lam^64 (re, im)
__global__ __launch_bounds__(64) void s4_precompute(const float* __restrict__ A_re,
                                                    const float* __restrict__ A_im,
                                                    const float* __restrict__ B_mat,
                                                    const float* __restrict__ C_mat,
                                                    const float* __restrict__ delta,
                                                    f16* __restrict__ Scat,
                                                    f16* __restrict__ Mcat,
                                                    float* __restrict__ lam64) {
  __shared__ float sdre[64], sdim[64], scr[64], sci[64], sK[64];
  const int d = blockIdx.x;
  const int n = threadIdx.x;
  const size_t off = (size_t)d * NST + n;
  const float are = A_re[off], aim = A_im[off];
  const float Bv = B_mat[off], Cv = C_mat[off];
  const float dt = delta[d];
  const float dre = dt * are, dim = dt * aim;
  // lambda = exp(dA)
  const float er = __expf(dre);
  const float lre = er * __cosf(dim), lim = er * __sinf(dim);
  // Bbar = (lambda - 1)/A * B
  const float inv = 1.0f / (are * are + aim * aim);
  const float br = ((lre - 1.0f) * are + lim * aim) * inv * Bv;
  const float bi = (lim * are - (lre - 1.0f) * aim) * inv * Bv;
  const float cr = Cv * br, ci = Cv * bi;   // coef = C * Bbar
  // lambda^64
  {
    const float e64 = __expf(64.0f * dre);
    lam64[(size_t)d * 128 + n * 2 + 0] = e64 * __cosf(64.0f * dim);
    lam64[(size_t)d * 128 + n * 2 + 1] = e64 * __sinf(64.0f * dim);
  }
  sdre[n] = dre; sdim[n] = dim; scr[n] = cr; sci[n] = ci;

  f16* gS = Scat + (size_t)d * 128 * NST;
  f16* gM = Mcat + (size_t)d * LC * KDIM;
  // powers p = 0..64 : lam^p = exp(p*dre)*(cos,sin)(p*dim)
  for (int p = 0; p <= 64; ++p) {
    const float fp = (float)p;
    const float ep = __expf(fp * dre);
    const float pr = ep * __cosf(fp * dim), pi = ep * __sinf(fp * dim);
    if (p <= 63) {                       // Sc[n][t] = lam^{63-t} * Bbar, t = 63-p
      const int t = 63 - p;
      gS[n * NST + t]          = (f16)(pr * br - pi * bi);
      gS[(64 + n) * NST + t]   = (f16)(pr * bi + pi * br);
    }
    if (p >= 1) {                        // O[t][n] = C * lam^{t+1}, t = p-1
      const int t = p - 1;
      gM[t * KDIM + 64 + n]  = (f16)(Cv * pr);
      gM[t * KDIM + 128 + n] = (f16)(-Cv * pi);
    }
  }
  __syncthreads();
  // K[j] = Re(sum_n coef_n * lam_n^j), j = this thread's index
  {
    float acc = 0.0f;
    const float fj = (float)n;
    for (int m = 0; m < 64; ++m) {
      const float ep = __expf(fj * sdre[m]);
      const float pr = ep * __cosf(fj * sdim[m]), pi = ep * __sinf(fj * sdim[m]);
      acc += scr[m] * pr - sci[m] * pi;
    }
    sK[n] = acc;
  }
  __syncthreads();
  // T0 row t = n
  for (int s = 0; s < 64; ++s)
    gM[n * KDIM + s] = (s <= n) ? (f16)sK[n - s] : (f16)0.0f;
}

// ---- Kernel 2: fused per-d chunked scan with WMMA GEMMs ----
// Per chunk I (operands f16, padded to N=16):
//   waves 0..3 : y(64x8)  = Mcat(64x192) @ [u_chunk; h_re; h_im] -> y_t global
//   waves 4..7 : v(128x8) = Scat(128x64) @ u_chunk               -> sV (LDS)
//   then all threads: h = lam^64 * h + v (complex, f32, in LDS)
// u chunk I+1 is prefetched into the other sRu buffer by wave 0 via async
// global->LDS copies while chunk I computes.
__global__ __launch_bounds__(256) void s4_main(const f16* __restrict__ u16,
                                               const f16* __restrict__ Scat,
                                               const f16* __restrict__ Mcat,
                                               const float* __restrict__ lam64,
                                               float* __restrict__ y_t) {
  __shared__ f16  sM[64 * KDIM];      // 24 KB
  __shared__ f16  sS[128 * NST];      // 16 KB
  __shared__ f16  sRu[2][16 * 64];    // u part of RHS [n][k], double buffered, 4 KB
  __shared__ f16  sRh[16 * 128];      // state part of RHS [n][k], 4 KB
  __shared__ float sH[128 * 8];       // state (re rows 0..63, im rows 64..127)
  __shared__ float sV[128 * 8];       // injection

  const int d = blockIdx.x;
  const int tid = threadIdx.x;
  const int wave = tid >> 5, lane = tid & 31;

  const f16* gUc = u16 + (size_t)d * NCH * 512;   // [chunk][b][64], 512 f16/chunk

#if HAVE_ASYNC_LDS
  if (wave == 0) async_copy_1k(gUc, &sRu[0][0], lane);   // prefetch chunk 0
#endif

  const f16* gM = Mcat + (size_t)d * 64 * KDIM;
  for (int i = tid; i < 64 * KDIM; i += 256) sM[i] = gM[i];
  const f16* gS = Scat + (size_t)d * 128 * NST;
  for (int i = tid; i < 128 * NST; i += 256) sS[i] = gS[i];
  for (int i = tid; i < 128 * 8; i += 256) sH[i] = 0.0f;
  // zero pad rows (n = 8..15) of both RHS pieces (u rows 0..7 rewritten per chunk)
  for (int i = tid; i < 2 * 16 * 64; i += 256) sRu[i >> 10][i & 1023] = (f16)0.0f;
  for (int i = tid; i < 16 * 128; i += 256) sRh[i] = (f16)0.0f;

  // scan coefficients for this thread's two (n,b) pairs: p0 = tid, p1 = tid+256
  const int p1 = tid + 256;
  const float a0 = lam64[(size_t)d * 128 + (tid >> 3) * 2 + 0];
  const float c0 = lam64[(size_t)d * 128 + (tid >> 3) * 2 + 1];
  const float a1 = lam64[(size_t)d * 128 + (p1 >> 3) * 2 + 0];
  const float c1 = lam64[(size_t)d * 128 + (p1 >> 3) * 2 + 1];

  float* gY = y_t + (size_t)d * SEQL * 8;

#if HAVE_ASYNC_LDS
  if (wave == 0) wait_async0();
#endif
  __syncthreads();

  for (int I = 0; I < NCH; ++I) {
    const int buf = I & 1;
#if HAVE_ASYNC_LDS
    // prefetch next chunk into the other buffer (its readers finished in I-1)
    if (wave == 0 && (I + 1) < NCH)
      async_copy_1k(gUc + (size_t)(I + 1) * 512, &sRu[buf ^ 1][0], lane);
#else
    // synchronous staging: layout matches global, straight copy
    sRu[buf][tid]       = gUc[(size_t)I * 512 + tid];
    sRu[buf][tid + 256] = gUc[(size_t)I * 512 + tid + 256];
#endif
    // stage state: sRh[b][r] = f16(h[r][b])  (1024 elems)
    for (int e = tid; e < 1024; e += 256) {
      const int bb = e >> 7, r = e & 127;
      sRh[bb * 128 + r] = (f16)sH[r * 8 + bb];
    }
    __syncthreads();

    if (wave < 4) {
      // output GEMM: mtile = wave, K = 192 (2 steps over u, 4 over state)
      v8f acc = {};
#pragma unroll
      for (int ks = 0; ks < 2; ++ks) {
        v16h a = load_frag_a(sM, KDIM, wave, ks, lane);
        v16h b = load_frag_b(&sRu[buf][0], 64, ks, lane);
        acc = __builtin_amdgcn_wmma_f32_16x16x32_f16(false, a, false, b,
                                                     (short)0, acc, false, false);
      }
#pragma unroll
      for (int ks = 2; ks < 6; ++ks) {
        v16h a = load_frag_a(sM, KDIM, wave, ks, lane);
        v16h b = load_frag_b(sRh, 128, ks - 2, lane);
        acc = __builtin_amdgcn_wmma_f32_16x16x32_f16(false, a, false, b,
                                                     (short)0, acc, false, false);
      }
      const int n = lane & 15, h = lane >> 4;
      if (n < 8) {
#pragma unroll
        for (int r = 0; r < 8; ++r) {
          const int m = r + 8 * h;
          gY[(I * 64 + wave * 16 + m) * 8 + n] = acc[r];
        }
      }
    } else {
      // state-injection GEMM: wave q handles v mtiles 2q, 2q+1; K = 64
      const int q = wave - 4;
#pragma unroll
      for (int mt = 0; mt < 2; ++mt) {
        const int mtile = 2 * q + mt;
        v8f acc = {};
#pragma unroll
        for (int ks = 0; ks < 2; ++ks) {
          v16h a = load_frag_a(sS, NST, mtile, ks, lane);
          v16h b = load_frag_b(&sRu[buf][0], 64, ks, lane);
          acc = __builtin_amdgcn_wmma_f32_16x16x32_f16(false, a, false, b,
                                                       (short)0, acc, false, false);
        }
        const int n = lane & 15, h = lane >> 4;
        if (n < 8) {
#pragma unroll
          for (int r = 0; r < 8; ++r) {
            const int m = r + 8 * h;
            sV[(mtile * 16 + m) * 8 + n] = acc[r];
          }
        }
      }
    }
    __syncthreads();

    // scan update: h = lam^64 * h + v  (complex per (n,b))
    {
      int n = tid >> 3, bb = tid & 7;
      float hr = sH[n * 8 + bb], hi = sH[(n + 64) * 8 + bb];
      float vr = sV[n * 8 + bb], vi = sV[(n + 64) * 8 + bb];
      sH[n * 8 + bb]        = a0 * hr - c0 * hi + vr;
      sH[(n + 64) * 8 + bb] = c0 * hr + a0 * hi + vi;
      n = p1 >> 3; bb = p1 & 7;
      hr = sH[n * 8 + bb]; hi = sH[(n + 64) * 8 + bb];
      vr = sV[n * 8 + bb]; vi = sV[(n + 64) * 8 + bb];
      sH[n * 8 + bb]        = a1 * hr - c1 * hi + vr;
      sH[(n + 64) * 8 + bb] = c1 * hr + a1 * hi + vi;
    }
#if HAVE_ASYNC_LDS
    if (wave == 0) wait_async0();    // next chunk's DMA landed before release
#endif
    __syncthreads();
  }
}

// ---- Kernel 3: y_t [d][l][b] f32 -> out (B,L,D) f32 ----
__global__ __launch_bounds__(256) void s4_transpose_y(const float* __restrict__ y_t,
                                                      float* __restrict__ out) {
  __shared__ float tile[32 * 32 * 8];          // [j][(i*8+b)], 32 KB
  const int l0 = blockIdx.x * 32;
  const int d0 = blockIdx.y * 32;
  const int tid = threadIdx.x;
  {
    const int j = tid >> 3;
    const int s = (tid & 7) * 32;
    size_t base = ((size_t)(d0 + j) * SEQL + l0) * 8 + s;
#pragma unroll
    for (int w = 0; w < 32; ++w)               // 128B contiguous load per thread
      tile[j * 256 + s + w] = y_t[base + w];
  }
  __syncthreads();
  const int b = tid >> 5, i = tid & 31;
  size_t base = ((size_t)b * SEQL + (l0 + i)) * DCH + d0;
#pragma unroll
  for (int j = 0; j < 32; ++j)                 // 128B contiguous store per thread
    out[base + j] = tile[j * 256 + i * 8 + b];
}

extern "C" void kernel_launch(void* const* d_in, const int* in_sizes, int n_in,
                              void* d_out, int out_size, void* d_ws, size_t ws_size,
                              hipStream_t stream) {
  (void)in_sizes; (void)n_in; (void)out_size; (void)ws_size;
  const float* u     = (const float*)d_in[0];
  const float* A_re  = (const float*)d_in[1];
  const float* A_im  = (const float*)d_in[2];
  const float* B_mat = (const float*)d_in[3];
  const float* C_mat = (const float*)d_in[4];
  const float* delta = (const float*)d_in[5];
  float* out = (float*)d_out;

  char* ws = (char*)d_ws;
  size_t off = 0;
  auto carve = [&](size_t bytes) -> void* {
    void* p = ws + off;
    off = (off + bytes + 255) & ~(size_t)255;
    return p;
  };
  f16*   u16   = (f16*)  carve((size_t)DCH * SEQL * 8 * sizeof(f16));     // 25 MB
  f16*   Scat  = (f16*)  carve((size_t)DCH * 128 * NST * sizeof(f16));    // 12.6 MB
  f16*   Mcat  = (f16*)  carve((size_t)DCH * LC * KDIM * sizeof(f16));    // 18.9 MB
  float* lam64 = (float*)carve((size_t)DCH * NST * 2 * sizeof(float));    // 0.4 MB
  float* y_t   = (float*)carve((size_t)DCH * SEQL * 8 * sizeof(float));   // 50 MB

  s4_transpose_u<<<dim3(SEQL / 32, DCH / 32), 256, 0, stream>>>(u, u16);
  s4_precompute<<<DCH, 64, 0, stream>>>(A_re, A_im, B_mat, C_mat, delta, Scat, Mcat, lam64);
  s4_main<<<DCH, 256, 0, stream>>>(u16, Scat, Mcat, lam64, y_t);
  s4_transpose_y<<<dim3(SEQL / 32, DCH / 32), 256, 0, stream>>>(y_t, out);
}
